// CorrCosine_88072599371882
// MI455X (gfx1250) — compile-verified, matrix operand order checked by reading, not run on hardware
//
#include <hip/hip_runtime.h>
#include <hip/hip_bf16.h>

typedef __attribute__((ext_vector_type(16))) _Float16 v16h;
typedef __attribute__((ext_vector_type(8)))  _Float16 v8h;
typedef __attribute__((ext_vector_type(8)))  float    v8f;

#define C_DIM 256
#define HW    4096          // 64*64 pixels per image
#define BM    128
#define BN    128
#define BK    32
#define LDSW  40            // BK + 8 halves pad (row stride 80B, 16B-aligned chunks)

// ---------------- Kernel 1: per-pixel channel rsqrt-norm ----------------
__global__ __launch_bounds__(256)
void corr_norm_kernel(const float* __restrict__ x, float* __restrict__ rn) {
    int p = blockIdx.x * 256 + threadIdx.x;     // 0 .. 8*4096-1
    int b = p >> 12;
    int m = p & (HW - 1);
    const float* xp = x + (size_t)b * C_DIM * HW + m;
    float s = 0.f;
#pragma unroll 8
    for (int c = 0; c < C_DIM; ++c) {
        float v = xp[(size_t)c * HW];
        s += v * v;
    }
    float n = sqrtf(s);
    rn[p] = 1.0f / fmaxf(n, 1e-12f);
}

// ---------------- Kernel 2: per-batch normalized-correlation GEMM ----------------
// A = cur (M side), B = ref (N side), K = channels. f16 WMMA, f32 accum.
__global__ __launch_bounds__(256)
void corr_gemm_kernel(const float* __restrict__ cur, const float* __restrict__ ref,
                      const float* __restrict__ rn_cur, const float* __restrict__ rn_ref,
                      float* __restrict__ out) {
    __shared__ _Float16 As[BM][LDSW];   // [m][k] (transposed at staging time)
    __shared__ _Float16 Bs[BN][LDSW];   // [n][k]

    const int b    = blockIdx.z;
    const int m0   = blockIdx.y * BM;
    const int n0   = blockIdx.x * BN;
    const int tid  = threadIdx.x;
    const int lane = tid & 31;
    const int wave = tid >> 5;          // 8 waves
    const int wm   = (wave >> 2) * 64;  // 2 wave rows  -> 64 M per wave
    const int wn   = (wave & 3) * 32;   // 4 wave cols  -> 32 N per wave
    const int lr   = lane & 15;
    const int kh8  = (lane >> 4) * 8;   // lanes 16-31 take the +8 K-halves

    // loader mapping: 128 m-columns x 2 k-halves of 16
    const int mm = tid & 127;
    const int kh = tid >> 7;            // 0 or 1
    const float ra = rn_cur[b * HW + m0 + mm];
    const float rb = rn_ref[b * HW + n0 + mm];

    const size_t binoff = (size_t)b * C_DIM * HW;
    const float* curb = cur + binoff;
    const float* refb = ref + binoff;

    v8f acc[4][2] = {};                 // 4 M-tiles x 2 N-tiles of 16x16

    for (int k0 = 0; k0 < C_DIM; k0 += BK) {
        // ---- stage: global f32 -> scale -> f16 -> LDS (transposed [m][k]) ----
        const float* pa = curb + (size_t)(k0 + kh * 16) * HW + (m0 + mm);
        const float* pb = refb + (size_t)(k0 + kh * 16) * HW + (n0 + mm);
        v8h a0, a1, b0, b1;
#pragma unroll
        for (int j = 0; j < 8; ++j) {
            a0[j] = (_Float16)(pa[(size_t)j * HW] * ra);
            a1[j] = (_Float16)(pa[(size_t)(j + 8) * HW] * ra);
            b0[j] = (_Float16)(pb[(size_t)j * HW] * rb);
            b1[j] = (_Float16)(pb[(size_t)(j + 8) * HW] * rb);
        }
        *(v8h*)&As[mm][kh * 16]     = a0;
        *(v8h*)&As[mm][kh * 16 + 8] = a1;
        *(v8h*)&Bs[mm][kh * 16]     = b0;
        *(v8h*)&Bs[mm][kh * 16 + 8] = b1;

        if (k0 + BK < C_DIM) {          // pull next K-panel toward L2/L0
            __builtin_prefetch(pa + (size_t)BK * HW, 0, 0);
            __builtin_prefetch(pb + (size_t)BK * HW, 0, 0);
        }
        __syncthreads();

        // ---- fragments from LDS (two b128 loads each, per 16-bit A layout) ----
        v16h af[4], bf[2];
#pragma unroll
        for (int mt = 0; mt < 4; ++mt) {
            const _Float16* r = &As[wm + mt * 16 + lr][0];
            v8h lo = *(const v8h*)(r + kh8);        // K = kh8 .. kh8+7
            v8h hi = *(const v8h*)(r + kh8 + 16);   // K = kh8+16 .. kh8+23
            af[mt] = __builtin_shufflevector(lo, hi, 0,1,2,3,4,5,6,7,8,9,10,11,12,13,14,15);
        }
#pragma unroll
        for (int nt = 0; nt < 2; ++nt) {
            const _Float16* r = &Bs[wn + nt * 16 + lr][0];
            v8h lo = *(const v8h*)(r + kh8);
            v8h hi = *(const v8h*)(r + kh8 + 16);
            bf[nt] = __builtin_shufflevector(lo, hi, 0,1,2,3,4,5,6,7,8,9,10,11,12,13,14,15);
        }

        // ---- 8 WMMAs: D = A x B + C ----
#pragma unroll
        for (int mt = 0; mt < 4; ++mt)
#pragma unroll
            for (int nt = 0; nt < 2; ++nt)
                acc[mt][nt] = __builtin_amdgcn_wmma_f32_16x16x32_f16(
                    false, af[mt], false, bf[nt],
                    (short)0, acc[mt][nt], false, false);

        __syncthreads();
    }

    // ---- epilogue: C/D layout -> out[b][m][n], m from cur, n from ref ----
    float* outb = out + ((size_t)b << 24);          // 4096*4096 per batch
    const int rowadd = (lane >> 4) * 8;             // lanes 16-31: M += 8
#pragma unroll
    for (int mt = 0; mt < 4; ++mt) {
#pragma unroll
        for (int nt = 0; nt < 2; ++nt) {
            const int gn = n0 + wn + nt * 16 + lr;
#pragma unroll
            for (int v = 0; v < 8; ++v) {
                const int gm = m0 + wm + mt * 16 + rowadd + v;
                outb[(size_t)gm * HW + gn] = acc[mt][nt][v];
            }
        }
    }
}

extern "C" void kernel_launch(void* const* d_in, const int* in_sizes, int n_in,
                              void* d_out, int out_size, void* d_ws, size_t ws_size,
                              hipStream_t stream) {
    // setup_inputs order: d_in[0] = ref_features, d_in[1] = cur_features (both f32)
    const float* ref = (const float*)d_in[0];
    const float* cur = (const float*)d_in[1];
    float* out = (float*)d_out;

    // workspace: 2 * 8*4096 f32 = 256 KB of rsqrt-norms
    float* rn_ref = (float*)d_ws;
    float* rn_cur = rn_ref + 8 * HW;

    corr_norm_kernel<<<dim3((8 * HW) / 256), 256, 0, stream>>>(ref, rn_ref);
    corr_norm_kernel<<<dim3((8 * HW) / 256), 256, 0, stream>>>(cur, rn_cur);

    // grid: 32x32 tiles of 128x128 per batch, 8 batches
    corr_gemm_kernel<<<dim3(HW / BN, HW / BM, 8), 256, 0, stream>>>(
        cur, ref, rn_cur, rn_ref, out);
}
